// SequenceTransformer_27968827032126
// MI455X (gfx1250) — matrix operand order repaired and checked by reading, not compiled, not run on hardware
//
#include <hip/hip_runtime.h>
#include <cstdint>
#include <cstddef>

// ---------------------------------------------------------------------------
// MI455X (gfx1250) implementation of the seq2seq Transformer reference.
// f32 residual stream (LN numerics); bf16 operands for every matmul so all
// matmul FLOPs run on v_wmma_f32_16x16x32_bf16 (wave32, f32 accumulate).
// bf16 weights (~88MB) stay L2-resident (192MB L2); activation traffic is
// the roofline term at 23.3 TB/s, so GEMM tiles are staged into LDS with
// GLOBAL_LOAD_ASYNC_TO_LDS_B128 (ASYNCcnt) and double-buffered; attention is
// flash-style so scores never touch HBM.
// ---------------------------------------------------------------------------

typedef __bf16 bf16;
typedef __attribute__((ext_vector_type(16))) __bf16 v16bf;
typedef __attribute__((ext_vector_type(8)))  __bf16 v8bf;
typedef __attribute__((ext_vector_type(8)))  float  v8f;

#define D_MODEL  512
#define N_HEAD   8
#define HEAD_DIM 64
#define SEQ      1024
#define BATCH    8
#define TOKENS   (BATCH * SEQ)   // 8192
#define D_FF     2048
#define WINDOW   64
#define LOG2E    1.44269504088896f

union Frag16 { v16bf v; v8bf h[2]; bf16 e[16]; };

__device__ __forceinline__ v8f wmma_bf16(v16bf a, v16bf b, v8f c) {
  // D = A(16x32) * B(32x16) + C, f32 accumulate. 8-arg form:
  // (neg_a, A, neg_b, B, c_mod, C, reuse_a, reuse_b)
  return __builtin_amdgcn_wmma_f32_16x16x32_bf16(false, a, false, b,
                                                 (short)0, c, false, false);
}

// CDNA5 async global->LDS DMA (per-lane, tracked by ASYNCcnt).
__device__ __forceinline__ void async_b128(void* lds, const void* g) {
  asm volatile("global_load_async_to_lds_b128 %0, %1, off"
               :: "v"((uint32_t)(uintptr_t)lds), "v"(g)
               : "memory");
}
__device__ __forceinline__ void wait_async0() {
  asm volatile("s_wait_asynccnt 0x0" ::: "memory");
}

// ---------------------------------------------------------------------------
// f32 -> bf16 cast (weights; grid-stride)
// ---------------------------------------------------------------------------
__global__ void k_cast(const float* __restrict__ in, bf16* __restrict__ out,
                       size_t n) {
  size_t i = (size_t)blockIdx.x * blockDim.x + threadIdx.x;
  size_t stride = (size_t)gridDim.x * blockDim.x;
  for (; i < n; i += stride) out[i] = (bf16)in[i];
}

// ---------------------------------------------------------------------------
// Input embedding: X[b,t,:] = src[b,t,:] @ W_in^T + b_in + pos[t,:]
// ---------------------------------------------------------------------------
__global__ __launch_bounds__(256)
void k_embed(const float* __restrict__ src, const float* __restrict__ Win,
             const float* __restrict__ bin, const float* __restrict__ pos,
             float* __restrict__ Xf, bf16* __restrict__ Xb) {
  const int idx = blockIdx.x * 256 + threadIdx.x;   // over TOKENS*512
  const int d  = idx & (D_MODEL - 1);
  const int bt = idx >> 9;
  const int t  = bt & (SEQ - 1);
  const float* sp = src + (size_t)bt * 64;
  const float* wp = Win + (size_t)d * 64;
  float acc = bin[d] + pos[(size_t)t * D_MODEL + d];
#pragma unroll
  for (int c = 0; c < 64; ++c) acc += sp[c] * wp[c];
  Xf[idx] = acc;
  Xb[idx] = (bf16)acc;
}

// ---------------------------------------------------------------------------
// Tiled WMMA GEMM:  C[M,N] = A[M,K](bf16) @ W[N,K]^T(bf16) + bias, opt ReLU.
// Block = 256 threads (8 waves), 128x64 output tile; each wave owns a 32x32
// register tile (4 accumulators -> 4 WMMAs per 32-deep K chunk). Tiles are
// staged via async global->LDS DMA, double buffered; s_wait_asynccnt + split
// barrier gate each buffer. M%128==0, N%64==0, K%32==0 hold for all GEMMs.
// ---------------------------------------------------------------------------
__global__ __launch_bounds__(256)
void k_gemm(const bf16* __restrict__ A, const bf16* __restrict__ W,
            const float* __restrict__ bias,
            float* __restrict__ Cf, bf16* __restrict__ Cb,
            int M, int N, int K, int relu) {
  __shared__ alignas(32) bf16 As[2][128][32];
  __shared__ alignas(32) bf16 Ws[2][64][32];
  const int tid  = threadIdx.x;
  const int lane = tid & 31;
  const int wave = tid >> 5;
  const int mi2 = wave >> 1;          // 0..3 (32-row group)
  const int nh  = wave & 1;           // 0..1 (32-col half)
  const int r  = lane & 15;
  const int hi = lane >> 4;
  const int m0 = blockIdx.y * 128;
  const int n0 = blockIdx.x * 64;

  // global->LDS staging assignment: A 128x32 (32B/thread), W 64x32 (16B/thread)
  const int arow = tid >> 1, acol = (tid & 1) << 4;
  const int wrow = tid >> 2, wcol = (tid & 3) << 3;
  const bf16* Ag = A + (size_t)(m0 + arow) * K + acol;
  const bf16* Wg = W + (size_t)(n0 + wrow) * K + wcol;

  v8f c00 = {0,0,0,0,0,0,0,0}, c01 = {0,0,0,0,0,0,0,0};
  v8f c10 = {0,0,0,0,0,0,0,0}, c11 = {0,0,0,0,0,0,0,0};

  const int nK = K >> 5;
  async_b128(&As[0][arow][acol],     Ag);
  async_b128(&As[0][arow][acol + 8], Ag + 8);
  async_b128(&Ws[0][wrow][wcol],     Wg);

  for (int kc = 0; kc < nK; ++kc) {
    wait_async0();                     // my DMAs into 'cur' have landed
    __syncthreads();                   // everyone's have; prior compute done
    const int cur = kc & 1;
    if (kc + 1 < nK) {
      const bf16* an = Ag + (size_t)(kc + 1) * 32;
      const bf16* wn = Wg + (size_t)(kc + 1) * 32;
      if (kc + 2 < nK) {               // global_prefetch_b8 for the K+2 tile
        __builtin_prefetch(an + 32, 0, 1);
        __builtin_prefetch(wn + 32, 0, 1);
      }
      async_b128(&As[cur ^ 1][arow][acol],     an);
      async_b128(&As[cur ^ 1][arow][acol + 8], an + 8);
      async_b128(&Ws[cur ^ 1][wrow][wcol],     wn);
    }
    // A fragments (16x32): lane(r,hi) holds K = 8*hi+0..7 and 16+8*hi+0..7
    Frag16 a0, a1;
    {
      const bf16* ar0 = &As[cur][mi2 * 32 + r][0];
      const bf16* ar1 = &As[cur][mi2 * 32 + 16 + r][0];
      a0.h[0] = *(const v8bf*)(ar0 + 8 * hi);
      a0.h[1] = *(const v8bf*)(ar0 + 16 + 8 * hi);
      a1.h[0] = *(const v8bf*)(ar1 + 8 * hi);
      a1.h[1] = *(const v8bf*)(ar1 + 16 + 8 * hi);
    }
    // B fragments (32x16): lane col = r, rows K = 8*hi+0..7 and 16+8*hi+0..7
    Frag16 b0f, b1f;
    b0f.h[0] = *(const v8bf*)(&Ws[cur][nh * 32 + r][8 * hi]);
    b0f.h[1] = *(const v8bf*)(&Ws[cur][nh * 32 + r][16 + 8 * hi]);
    b1f.h[0] = *(const v8bf*)(&Ws[cur][nh * 32 + 16 + r][8 * hi]);
    b1f.h[1] = *(const v8bf*)(&Ws[cur][nh * 32 + 16 + r][16 + 8 * hi]);
    v16bf b0 = b0f.v;
    v16bf b1 = b1f.v;
    c00 = wmma_bf16(a0.v, b0, c00);
    c01 = wmma_bf16(a0.v, b1, c01);
    c10 = wmma_bf16(a1.v, b0, c10);
    c11 = wmma_bf16(a1.v, b1, c11);
  }

  // Epilogue. C/D layout: element v of lane(r,hi) -> row v+8*hi, col r.
  const int n_base = n0 + nh * 32 + r;
  const float bi0 = bias[n_base], bi1 = bias[n_base + 16];
#pragma unroll
  for (int v = 0; v < 8; ++v) {
    const int mA = m0 + mi2 * 32 + 8 * hi + v;
    const int mB = mA + 16;
    float x00 = c00[v] + bi0, x01 = c01[v] + bi1;
    float x10 = c10[v] + bi0, x11 = c11[v] + bi1;
    if (relu) {
      x00 = fmaxf(x00, 0.f); x01 = fmaxf(x01, 0.f);
      x10 = fmaxf(x10, 0.f); x11 = fmaxf(x11, 0.f);
    }
    const size_t oA = (size_t)mA * N + n_base;
    const size_t oB = (size_t)mB * N + n_base;
    if (Cf) { Cf[oA] = x00; Cf[oA + 16] = x01; Cf[oB] = x10; Cf[oB + 16] = x11; }
    if (Cb) { Cb[oA] = (bf16)x00; Cb[oA + 16] = (bf16)x01;
              Cb[oB] = (bf16)x10; Cb[oB + 16] = (bf16)x11; }
  }
}

// ---------------------------------------------------------------------------
// Flash attention, one wave per (b, head, 16-query tile).
// Scores: 2 chained WMMAs per 16-key column tile (head_dim 64 reduction).
// V tile is DMA'd to LDS asynchronously and consumed after the softmax
// (overlapping DMA with score WMMAs). P restaged via LDS (C->A layout);
// 4 P@V WMMAs per 32-key tile. window<0 => dense cross-attention.
// ---------------------------------------------------------------------------
__global__ __launch_bounds__(32)
void k_attn(const bf16* __restrict__ Q, int qs,
            const bf16* __restrict__ Kt, int ks,
            const bf16* __restrict__ V, int vs,
            bf16* __restrict__ O, int os,
            int Sq, int Sk, int window) {
  __shared__ alignas(32) bf16 Ps[16][32];
  __shared__ alignas(32) bf16 Vs[32][64];
  const int lane = threadIdx.x;
  const int r  = lane & 15;
  const int hi = lane >> 4;
  const int q0 = blockIdx.x * 16;
  const int h  = blockIdx.y;
  const int b  = blockIdx.z;

  const bf16* Qb = Q  + (size_t)b * Sq * qs + h * HEAD_DIM;
  const bf16* Kb = Kt + (size_t)b * Sk * ks + h * HEAD_DIM;
  const bf16* Vb = V  + (size_t)b * Sk * vs + h * HEAD_DIM;

  // Q tile (16x64) as two A fragments (d 0..31, 32..63)
  Frag16 a0, a1;
  {
    const bf16* qr = Qb + (size_t)(q0 + r) * qs;
    a0.h[0] = *(const v8bf*)(qr + 16 * hi);
    a0.h[1] = *(const v8bf*)(qr + 16 * hi + 8);
    a1.h[0] = *(const v8bf*)(qr + 32 + 16 * hi);
    a1.h[1] = *(const v8bf*)(qr + 40 + 16 * hi);
  }

  float mrun[8], lrun[8];
  v8f oacc[4];
#pragma unroll
  for (int v = 0; v < 8; ++v) { mrun[v] = -3e38f; lrun[v] = 0.f; }
#pragma unroll
  for (int c = 0; c < 4; ++c)
#pragma unroll
    for (int v = 0; v < 8; ++v) oacc[c][v] = 0.f;

  int lo, hiE;
  if (window >= 0) {
    lo = q0 - window; if (lo < 0) lo = 0; lo &= ~31;
    hiE = (q0 + 16 + window + 31) & ~31; if (hiE > Sk) hiE = Sk;
  } else { lo = 0; hiE = Sk; }

  for (int kv = lo; kv < hiE; kv += 32) {
    // Kick off async DMA of V tile [32 x 64] (128B/lane) into LDS
    {
      const bf16* vr = Vb + (size_t)(kv + lane) * vs;
#pragma unroll
      for (int c8 = 0; c8 < 8; ++c8)
        async_b128(&Vs[lane][c8 * 8], vr + c8 * 8);
    }
    // Scores: S[16q x 32k] as two 16-col WMMA results (overlaps V DMA)
    v8f s0 = {0, 0, 0, 0, 0, 0, 0, 0};
    v8f s1 = {0, 0, 0, 0, 0, 0, 0, 0};
    {
      const bf16* kr0 = Kb + (size_t)(kv + r) * ks;        // B col = r
      v16bf kb0 = *(const v16bf*)(kr0 + 16 * hi);          // d rows 0..31
      v16bf kb1 = *(const v16bf*)(kr0 + 32 + 16 * hi);     // d rows 32..63
      s0 = wmma_bf16(a0.v, kb0, s0);
      s0 = wmma_bf16(a1.v, kb1, s0);
      const bf16* kr1 = Kb + (size_t)(kv + 16 + r) * ks;
      kb0 = *(const v16bf*)(kr1 + 16 * hi);
      kb1 = *(const v16bf*)(kr1 + 32 + 16 * hi);
      s1 = wmma_bf16(a0.v, kb0, s1);
      s1 = wmma_bf16(a1.v, kb1, s1);
    }
    // Scale + band mask + row max (16-lane shuffle reduction within hi-group)
    float p0[8], p1[8], mx[8];
#pragma unroll
    for (int v = 0; v < 8; ++v) {
      float x0 = s0[v] * 0.125f;                 // 1/sqrt(64)
      float x1 = s1[v] * 0.125f;
      if (window >= 0) {
        const int rowg = q0 + v + 8 * hi;
        int d0 = rowg - (kv + r);      if (d0 < 0) d0 = -d0;
        int d1 = rowg - (kv + 16 + r); if (d1 < 0) d1 = -d1;
        if (d0 > window) x0 = -1e30f;
        if (d1 > window) x1 = -1e30f;
      }
      p0[v] = x0; p1[v] = x1;
      mx[v] = fmaxf(x0, x1);
    }
#pragma unroll
    for (int v = 0; v < 8; ++v)
#pragma unroll
      for (int off = 1; off < 16; off <<= 1)
        mx[v] = fmaxf(mx[v], __shfl_xor(mx[v], off, 32));
    // Online-softmax rescale
    float corr[8];
#pragma unroll
    for (int v = 0; v < 8; ++v) {
      const float mnew = fmaxf(mrun[v], mx[v]);
      corr[v] = exp2f((mrun[v] - mnew) * LOG2E);
      mrun[v] = mnew;
      lrun[v] *= corr[v];
    }
#pragma unroll
    for (int c = 0; c < 4; ++c)
#pragma unroll
      for (int v = 0; v < 8; ++v) oacc[c][v] *= corr[v];
    // P = exp(s - m); stage bf16 P tile to LDS in row-major [16q x 32k]
    float lsum[8];
#pragma unroll
    for (int v = 0; v < 8; ++v) {
      const float e0 = exp2f((p0[v] - mrun[v]) * LOG2E);
      const float e1 = exp2f((p1[v] - mrun[v]) * LOG2E);
      Ps[v + 8 * hi][r]      = (bf16)e0;
      Ps[v + 8 * hi][16 + r] = (bf16)e1;
      lsum[v] = e0 + e1;
    }
#pragma unroll
    for (int v = 0; v < 8; ++v) {
#pragma unroll
      for (int off = 1; off < 16; off <<= 1)
        lsum[v] += __shfl_xor(lsum[v], off, 32);
      lrun[v] += lsum[v];
    }
    // P A-fragment (16q x 32kv); single wave => LDS ops are in-order
    Frag16 pa;
    {
      const bf16* pr = &Ps[r][0];
      pa.h[0] = *(const v8bf*)(pr + 16 * hi);
      pa.h[1] = *(const v8bf*)(pr + 16 * hi + 8);
    }
    wait_async0();                      // V tile DMA complete
    __syncthreads();
    // 4 output WMMAs: B = V 32(kv) x 16(d chunk); B elem e -> kv row e+16*hi
#pragma unroll
    for (int c = 0; c < 4; ++c) {
      Frag16 vbf;
#pragma unroll
      for (int e = 0; e < 16; ++e)
        vbf.e[e] = Vs[e + 16 * hi][c * 16 + r];
      oacc[c] = wmma_bf16(pa.v, vbf.v, oacc[c]);
    }
  }

  // Normalize and write bf16 heads-concat output (feeds out-proj GEMM)
  bf16* Ob = O + (size_t)b * Sq * os + h * HEAD_DIM;
#pragma unroll
  for (int v = 0; v < 8; ++v) {
    const float inv = 1.0f / lrun[v];
    const int row = q0 + v + 8 * hi;
#pragma unroll
    for (int c = 0; c < 4; ++c)
      Ob[(size_t)row * os + c * 16 + r] = (bf16)(oacc[c][v] * inv);
  }
}

// ---------------------------------------------------------------------------
// Fused residual-add + LayerNorm over D_MODEL=512. One block per token row.
// ---------------------------------------------------------------------------
__global__ __launch_bounds__(256)
void k_add_ln(const float* __restrict__ X, const float* __restrict__ Aadd,
              const float* __restrict__ g, const float* __restrict__ bta,
              float* __restrict__ Yf, bf16* __restrict__ Yb) {
  __shared__ float rsum[8], rsq[8];
  const int tid = threadIdx.x;
  const size_t base = (size_t)blockIdx.x * D_MODEL;
  float v0 = X[base + tid];
  float v1 = X[base + tid + 256];
  if (Aadd) { v0 += Aadd[base + tid]; v1 += Aadd[base + tid + 256]; }
  float s = v0 + v1, q = v0 * v0 + v1 * v1;
#pragma unroll
  for (int off = 1; off < 32; off <<= 1) {
    s += __shfl_xor(s, off, 32);
    q += __shfl_xor(q, off, 32);
  }
  if ((tid & 31) == 0) { rsum[tid >> 5] = s; rsq[tid >> 5] = q; }
  __syncthreads();
  float ts = 0.f, tq = 0.f;
#pragma unroll
  for (int w = 0; w < 8; ++w) { ts += rsum[w]; tq += rsq[w]; }
  const float mu  = ts * (1.0f / D_MODEL);
  const float var = tq * (1.0f / D_MODEL) - mu * mu;
  const float rs  = rsqrtf(var + 1e-5f);
  const float y0 = (v0 - mu) * rs * g[tid] + bta[tid];
  const float y1 = (v1 - mu) * rs * g[tid + 256] + bta[tid + 256];
  if (Yf) { Yf[base + tid] = y0; Yf[base + tid + 256] = y1; }
  if (Yb) { Yb[base + tid] = (bf16)y0; Yb[base + tid + 256] = (bf16)y1; }
}

// ---------------------------------------------------------------------------
// Host-side helpers
// ---------------------------------------------------------------------------
static void gemm(const bf16* A, const bf16* W, const float* bias,
                 float* Cf, bf16* Cb, int M, int N, int K, int relu,
                 hipStream_t s) {
  dim3 grid(N / 64, M / 128);
  k_gemm<<<grid, dim3(256), 0, s>>>(A, W, bias, Cf, Cb, M, N, K, relu);
}

static void attn(const bf16* Q, int qs, const bf16* K, int ks,
                 const bf16* V, int vs, bf16* O, int os,
                 int Sq, int Sk, int window, hipStream_t s) {
  dim3 grid(Sq / 16, N_HEAD, BATCH);
  k_attn<<<grid, dim3(32), 0, s>>>(Q, qs, K, ks, V, vs, O, os, Sq, Sk, window);
}

static void cast_w(const float* in, bf16* out, size_t n, hipStream_t s) {
  k_cast<<<dim3(2048), dim3(256), 0, s>>>(in, out, n);
}

static void add_ln(const float* X, const float* A, const float* g,
                   const float* bta, float* Yf, bf16* Yb, hipStream_t s) {
  k_add_ln<<<dim3(TOKENS), dim3(256), 0, s>>>(X, A, g, bta, Yf, Yb);
}

extern "C" void kernel_launch(void* const* d_in, const int* in_sizes, int n_in,
                              void* d_out, int out_size, void* d_ws,
                              size_t ws_size, hipStream_t stream) {
  (void)in_sizes; (void)n_in; (void)out_size; (void)ws_size;
  // setup_inputs() dict order (params dict flattened depth-first, insertion order)
  const float* src      = (const float*)d_in[0];
  const float* tgt      = (const float*)d_in[1];
  const float* W_in     = (const float*)d_in[2];
  const float* b_in     = (const float*)d_in[3];
  const float* pos      = (const float*)d_in[4];
  const float* e_qkv_w  = (const float*)d_in[5];
  const float* e_qkv_b  = (const float*)d_in[6];
  const float* e_o_w    = (const float*)d_in[7];
  const float* e_o_b    = (const float*)d_in[8];
  const float* e_ff1_w  = (const float*)d_in[9];
  const float* e_ff1_b  = (const float*)d_in[10];
  const float* e_ff2_w  = (const float*)d_in[11];
  const float* e_ff2_b  = (const float*)d_in[12];
  const float* e_ln1_g  = (const float*)d_in[13];
  const float* e_ln1_b  = (const float*)d_in[14];
  const float* e_ln2_g  = (const float*)d_in[15];
  const float* e_ln2_b  = (const float*)d_in[16];
  const float* enc_ng   = (const float*)d_in[17];
  const float* enc_nb   = (const float*)d_in[18];
  const float* sa_qkv_w = (const float*)d_in[19];
  const float* sa_qkv_b = (const float*)d_in[20];
  const float* sa_o_w   = (const float*)d_in[21];
  const float* sa_o_b   = (const float*)d_in[22];
  const float* ca_qkv_w = (const float*)d_in[23];
  const float* ca_qkv_b = (const float*)d_in[24];
  const float* ca_o_w   = (const float*)d_in[25];
  const float* ca_o_b   = (const float*)d_in[26];
  const float* d_ff1_w  = (const float*)d_in[27];
  const float* d_ff1_b  = (const float*)d_in[28];
  const float* d_ff2_w  = (const float*)d_in[29];
  const float* d_ff2_b  = (const float*)d_in[30];
  const float* d_ln1_g  = (const float*)d_in[31];
  const float* d_ln1_b  = (const float*)d_in[32];
  const float* d_ln2_g  = (const float*)d_in[33];
  const float* d_ln2_b  = (const float*)d_in[34];
  const float* d_ln3_g  = (const float*)d_in[35];
  const float* d_ln3_b  = (const float*)d_in[36];
  const float* dec_ng   = (const float*)d_in[37];
  const float* dec_nb   = (const float*)d_in[38];
  const float* W_out    = (const float*)d_in[39];
  const float* b_out    = (const float*)d_in[40];
  float* out = (float*)d_out;

  // Workspace bump allocator (~225 MB total)
  uint8_t* ws = (uint8_t*)d_ws;
  size_t off = 0;
  auto alloc = [&](size_t bytes) -> void* {
    void* p = ws + off;
    off = (off + bytes + 255) & ~(size_t)255;
    return p;
  };
  const size_t SQ  = (size_t)6 * 1536 * 512;  // per-stack qkv weights
  const size_t SO  = (size_t)6 * 512 * 512;
  const size_t SF1 = (size_t)6 * 2048 * 512;
  const size_t SF2 = (size_t)6 * 512 * 2048;

  bf16* w_eqkv = (bf16*)alloc(SQ * 2);
  bf16* w_eo   = (bf16*)alloc(SO * 2);
  bf16* w_ef1  = (bf16*)alloc(SF1 * 2);
  bf16* w_ef2  = (bf16*)alloc(SF2 * 2);
  bf16* w_sqkv = (bf16*)alloc(SQ * 2);
  bf16* w_so   = (bf16*)alloc(SO * 2);
  bf16* w_cqkv = (bf16*)alloc(SQ * 2);
  bf16* w_co   = (bf16*)alloc(SO * 2);
  bf16* w_df1  = (bf16*)alloc(SF1 * 2);
  bf16* w_df2  = (bf16*)alloc(SF2 * 2);
  bf16* w_out  = (bf16*)alloc((size_t)64 * 512 * 2);

  float* xf   = (float*)alloc((size_t)TOKENS * 512 * 4);
  bf16*  xb   = (bf16*) alloc((size_t)TOKENS * 512 * 2);
  float* yf   = (float*)alloc((size_t)TOKENS * 512 * 4);
  bf16*  yb   = (bf16*) alloc((size_t)TOKENS * 512 * 2);
  bf16*  memb = (bf16*) alloc((size_t)TOKENS * 512 * 2);
  bf16*  t0b  = (bf16*) alloc((size_t)TOKENS * 2048 * 2); // qkv / ffn hidden
  float* t1f  = (float*)alloc((size_t)TOKENS * 512 * 4);  // proj outputs
  bf16*  attb = (bf16*) alloc((size_t)TOKENS * 512 * 2);
  bf16*  qb   = (bf16*) alloc((size_t)TOKENS * 512 * 2);
  bf16*  kvb  = (bf16*) alloc((size_t)TOKENS * 1024 * 2);

  // Cast all weight matrices to bf16 once per call (L2-resident afterwards)
  cast_w(e_qkv_w,  w_eqkv, SQ,  stream);
  cast_w(e_o_w,    w_eo,   SO,  stream);
  cast_w(e_ff1_w,  w_ef1,  SF1, stream);
  cast_w(e_ff2_w,  w_ef2,  SF2, stream);
  cast_w(sa_qkv_w, w_sqkv, SQ,  stream);
  cast_w(sa_o_w,   w_so,   SO,  stream);
  cast_w(ca_qkv_w, w_cqkv, SQ,  stream);
  cast_w(ca_o_w,   w_co,   SO,  stream);
  cast_w(d_ff1_w,  w_df1,  SF1, stream);
  cast_w(d_ff2_w,  w_df2,  SF2, stream);
  cast_w(W_out,    w_out,  (size_t)64 * 512, stream);

  // Embeddings
  k_embed<<<dim3(TOKENS * 512 / 256), dim3(256), 0, stream>>>(
      src, W_in, b_in, pos, xf, xb);
  k_embed<<<dim3(TOKENS * 512 / 256), dim3(256), 0, stream>>>(
      tgt, W_in, b_in, pos, yf, yb);

  const int M = TOKENS;
  // ------------------------------ Encoder ------------------------------
  for (int l = 0; l < 6; ++l) {
    gemm(xb, w_eqkv + (size_t)l * 1536 * 512, e_qkv_b + (size_t)l * 1536,
         nullptr, t0b, M, 1536, 512, 0, stream);
    attn(t0b, 1536, t0b + 512, 1536, t0b + 1024, 1536,
         attb, 512, SEQ, SEQ, WINDOW, stream);
    gemm(attb, w_eo + (size_t)l * 512 * 512, e_o_b + (size_t)l * 512,
         t1f, nullptr, M, 512, 512, 0, stream);
    add_ln(xf, t1f, e_ln1_g + (size_t)l * 512, e_ln1_b + (size_t)l * 512,
           xf, xb, stream);
    gemm(xb, w_ef1 + (size_t)l * 2048 * 512, e_ff1_b + (size_t)l * 2048,
         nullptr, t0b, M, 2048, 512, 1, stream);
    gemm(t0b, w_ef2 + (size_t)l * 512 * 2048, e_ff2_b + (size_t)l * 512,
         t1f, nullptr, M, 512, 2048, 0, stream);
    add_ln(xf, t1f, e_ln2_g + (size_t)l * 512, e_ln2_b + (size_t)l * 512,
           xf, xb, stream);
  }
  add_ln(xf, nullptr, enc_ng, enc_nb, nullptr, memb, stream);  // encoder norm

  // ------------------------------ Decoder ------------------------------
  for (int l = 0; l < 6; ++l) {
    // Self-attention (banded)
    gemm(yb, w_sqkv + (size_t)l * 1536 * 512, sa_qkv_b + (size_t)l * 1536,
         nullptr, t0b, M, 1536, 512, 0, stream);
    attn(t0b, 1536, t0b + 512, 1536, t0b + 1024, 1536,
         attb, 512, SEQ, SEQ, WINDOW, stream);
    gemm(attb, w_so + (size_t)l * 512 * 512, sa_o_b + (size_t)l * 512,
         t1f, nullptr, M, 512, 512, 0, stream);
    add_ln(yf, t1f, d_ln1_g + (size_t)l * 512, d_ln1_b + (size_t)l * 512,
           yf, yb, stream);
    // Cross-attention (dense): q from y, k/v from encoder memory
    gemm(yb, w_cqkv + (size_t)l * 1536 * 512, ca_qkv_b + (size_t)l * 1536,
         nullptr, qb, M, 512, 512, 0, stream);
    gemm(memb, w_cqkv + (size_t)l * 1536 * 512 + (size_t)512 * 512,
         ca_qkv_b + (size_t)l * 1536 + 512,
         nullptr, kvb, M, 1024, 512, 0, stream);
    attn(qb, 512, kvb, 1024, kvb + 512, 1024,
         attb, 512, SEQ, SEQ, -1, stream);
    gemm(attb, w_co + (size_t)l * 512 * 512, ca_o_b + (size_t)l * 512,
         t1f, nullptr, M, 512, 512, 0, stream);
    add_ln(yf, t1f, d_ln2_g + (size_t)l * 512, d_ln2_b + (size_t)l * 512,
           yf, yb, stream);
    // FFN
    gemm(yb, w_df1 + (size_t)l * 2048 * 512, d_ff1_b + (size_t)l * 2048,
         nullptr, t0b, M, 2048, 512, 1, stream);
    gemm(t0b, w_df2 + (size_t)l * 512 * 2048, d_ff2_b + (size_t)l * 512,
         t1f, nullptr, M, 512, 2048, 0, stream);
    add_ln(yf, t1f, d_ln3_g + (size_t)l * 512, d_ln3_b + (size_t)l * 512,
           yf, yb, stream);
  }
  add_ln(yf, nullptr, dec_ng, dec_nb, nullptr, yb, stream);    // decoder norm

  // Output projection: [8192,512] @ [64,512]^T -> d_out (f32)
  gemm(yb, w_out, b_out, out, nullptr, M, 64, 512, 0, stream);
}
